// RGBBlinnPhongRendererWithVisibility_66116726554752
// MI455X (gfx1250) — matrix operand order repaired and checked by reading, not compiled, not run on hardware
//
#include <hip/hip_runtime.h>
#include <hip/hip_bf16.h>

typedef __attribute__((ext_vector_type(16))) _Float16 v16h;
typedef __attribute__((ext_vector_type(8)))  float    v8f;

#define SHADE_EPS 1e-6f

// ---------------------------------------------------------------------------
// Zero the per-ray accumulators in workspace: acc[R][4] = {sum w*r, w*g, w*b, sum w}
// ---------------------------------------------------------------------------
__global__ void zero_ws_kernel(float* __restrict__ acc, int n) {
    int i = blockIdx.x * blockDim.x + threadIdx.x;
    if (i < n) acc[i] = 0.0f;
}

// ---------------------------------------------------------------------------
// Per-light Blinn-Phong shading (all f32, matches reference math)
// ---------------------------------------------------------------------------
__device__ __forceinline__ void shade_light(
    float lx, float ly, float lz,
    float cr, float cg, float cb, float vis,
    float nx, float ny, float nz,
    float vx, float vy, float vz,
    float ar, float ag, float ab, float shin,
    float& r0, float& r1, float& r2)
{
    float hx = lx + vx, hy = ly + vy, hz = lz + vz;
    float inv = rsqrtf(hx * hx + hy * hy + hz * hz);
    float dnl = fminf(fmaxf(nx * lx + ny * ly + nz * lz, 0.0f), 1.0f);
    float dnh = fminf(fmaxf((nx * hx + ny * hy + nz * hz) * inv, 0.0f), 1.0f);
    // powf(max(dnh,eps), s) == exp2(s * log2(max(dnh,eps)))  (base > 0)
    float spec = exp2f(shin * log2f(fmaxf(dnh, SHADE_EPS)));
    r0 += (cr * vis) * (ar * dnl + spec);
    r1 += (cg * vis) * (ag * dnl + spec);
    r2 += (cb * vis) * (ab * dnl + spec);
}

// ---------------------------------------------------------------------------
// Shade one sample: stream its J lights with float4 (global_load_b128) loads.
// Per-thread light block is contiguous (768B dirs / 768B colors / 256B vis),
// 16B aligned, so every fetched line is fully consumed out of WGP$.
// NG = J/4 as a compile-time constant (0 => runtime ngroups).
// ---------------------------------------------------------------------------
template <int NG>
__device__ __forceinline__ void shade_sample_full(
    int t, int ngroups_rt,
    const float* __restrict__ albedos, const float* __restrict__ normals,
    const float* __restrict__ ldirs,   const float* __restrict__ lcols,
    const float* __restrict__ vis,     const float* __restrict__ shininess,
    const float* __restrict__ c2w,
    float& r0, float& r1, float& r2)
{
    // view_dir = c2w @ [0,0,-1,1] = col3 - col2
    float vx = c2w[(size_t)t * 12 + 3]  - c2w[(size_t)t * 12 + 2];
    float vy = c2w[(size_t)t * 12 + 7]  - c2w[(size_t)t * 12 + 6];
    float vz = c2w[(size_t)t * 12 + 11] - c2w[(size_t)t * 12 + 10];
    float nx = normals[(size_t)t * 3 + 0];
    float ny = normals[(size_t)t * 3 + 1];
    float nz = normals[(size_t)t * 3 + 2];
    float ar = albedos[(size_t)t * 3 + 0];
    float ag = albedos[(size_t)t * 3 + 1];
    float ab = albedos[(size_t)t * 3 + 2];
    float sh = shininess[t];
    r0 = 0.0f; r1 = 0.0f; r2 = 0.0f;

    const int ngroups = (NG > 0) ? NG : ngroups_rt;   // 4 lights per group
    const float4* d4 = (const float4*)ldirs + (size_t)t * (3 * ngroups);
    const float4* c4 = (const float4*)lcols + (size_t)t * (3 * ngroups);
    const float4* v4 = (const float4*)vis   + (size_t)t * ngroups;

#pragma unroll 4
    for (int g = 0; g < ngroups; ++g) {
        // prefetch next group's cache lines (global_prefetch_b8, speculative)
        __builtin_prefetch(&d4[3 * g + 3], 0, 0);
        __builtin_prefetch(&c4[3 * g + 3], 0, 0);

        float4 d0 = d4[3 * g + 0], d1 = d4[3 * g + 1], d2 = d4[3 * g + 2];
        float4 e0 = c4[3 * g + 0], e1 = c4[3 * g + 1], e2 = c4[3 * g + 2];
        float4 vv = v4[g];

        shade_light(d0.x, d0.y, d0.z, e0.x, e0.y, e0.z, vv.x,
                    nx, ny, nz, vx, vy, vz, ar, ag, ab, sh, r0, r1, r2);
        shade_light(d0.w, d1.x, d1.y, e0.w, e1.x, e1.y, vv.y,
                    nx, ny, nz, vx, vy, vz, ar, ag, ab, sh, r0, r1, r2);
        shade_light(d1.z, d1.w, d2.x, e1.z, e1.w, e2.x, vv.z,
                    nx, ny, nz, vx, vy, vz, ar, ag, ab, sh, r0, r1, r2);
        shade_light(d2.y, d2.z, d2.w, e2.y, e2.z, e2.w, vv.w,
                    nx, ny, nz, vx, vy, vz, ar, ag, ab, sh, r0, r1, r2);
    }
}

// ---------------------------------------------------------------------------
// Main shading kernel. Full blocks only (EXEC all-ones => WMMA legal).
// Wave-level segmented reduction via v_wmma_f32_16x16x32_f16:
//   D[slot][chan] = sum_lane A[slot][lane] * B[lane][chan]
//   A = exact 0/1 selector (f16), B = lane payload (w*r, w*g, w*b, w) in f16,
//   accumulation in f32 inside the WMMA. Rays are sorted, so a wave usually
//   spans 1-2 rays: one WMMA replaces ~128 colliding atomics with <= span*4.
// ---------------------------------------------------------------------------
template <int NG>
__global__ void __launch_bounds__(256) shade_kernel(
    const float* __restrict__ albedos, const float* __restrict__ normals,
    const float* __restrict__ ldirs,   const float* __restrict__ lcols,
    const float* __restrict__ vis,     const float* __restrict__ weights,
    const float* __restrict__ shininess, const float* __restrict__ c2w,
    const int* __restrict__ ray_idx,   float* __restrict__ acc,
    int ngroups_rt)
{
    __shared__ float sp[256 * 4];
    __shared__ int   sray[256];

    const int t = blockIdx.x * 256 + threadIdx.x;   // sample id (full blocks)

    float r0, r1, r2;
    shade_sample_full<NG>(t, ngroups_rt, albedos, normals, ldirs, lcols, vis,
                          shininess, c2w, r0, r1, r2);

    const float w   = weights[t];
    const int   ray = ray_idx[t];

    sp[threadIdx.x * 4 + 0] = w * r0;
    sp[threadIdx.x * 4 + 1] = w * r1;
    sp[threadIdx.x * 4 + 2] = w * r2;
    sp[threadIdx.x * 4 + 3] = w;
    sray[threadIdx.x] = ray;
    __syncthreads();

    const int lane = threadIdx.x & 31;
    const int wb   = threadIdx.x & ~31;         // wave base thread in block
    const int hi   = (lane >> 4) & 1;           // lane half (0: 0-15, 1: 16-31)
    const int M    = lane & 15;                 // A-matrix row owned by this lane pair
    const int col  = lane & 15;                 // D column read by this lane

    const int baseRay = sray[wb];               // rays sorted => wave min
    const int lastRay = sray[wb + 31];
    const int nslots  = lastRay - baseRay + 1;  // distinct-ray span of this wave

    // A: 16x32 f16, layout per ISA 7.12.2 (16-bit A 16x32):
    //   lanes 0-15 hold row M, K = {h:h<8} and {h+8:h>=8}; lanes 16-31: +8 / +16
    v16h av;
#pragma unroll
    for (int h = 0; h < 16; ++h) {
        int K = (h < 8) ? (h + 8 * hi) : ((h - 8) + 16 + 8 * hi);
        av[h] = (sray[wb + K] == baseRay + M) ? (_Float16)1.0f : (_Float16)0.0f;
    }

    // B: 32x16 f16, column = lane&15, K = {h+8*hi:h<8} and {h+8+8*hi:h>=8} (same as A);
    //   columns 0..3 carry (w*r, w*g, w*b, w) of thread (wb + K); others zero.
    v16h bv;
#pragma unroll
    for (int h = 0; h < 16; ++h) {
        int k = (h < 8) ? (h + 8 * hi) : (h + 8 + 8 * hi);
        bv[h] = (col < 4) ? (_Float16)sp[(wb + k) * 4 + col] : (_Float16)0.0f;
    }

    v8f cacc = {};
    cacc = __builtin_amdgcn_wmma_f32_16x16x32_f16(
        /*neg_a=*/false, av, /*neg_b=*/false, bv,
        /*c_mod=*/(short)0, cacc, /*reuse_a=*/false, /*reuse_b=*/false);

    if (nslots <= 16) {
        // C/D layout: lanes 0-15 VGPR r -> D[r][lane]; lanes 16-31 -> D[8+r][lane-16]
        if (col < 4) {
#pragma unroll
            for (int r = 0; r < 8; ++r) {
                int slot = r + 8 * hi;
                if (slot < nslots)
                    atomicAdd(&acc[(size_t)(baseRay + slot) * 4 + col], cacc[r]);
            }
        }
    } else {
        // rare: wave spans >16 rays -> direct per-lane atomics (exact f32 path)
        atomicAdd(&acc[(size_t)ray * 4 + 0], w * r0);
        atomicAdd(&acc[(size_t)ray * 4 + 1], w * r1);
        atomicAdd(&acc[(size_t)ray * 4 + 2], w * r2);
        atomicAdd(&acc[(size_t)ray * 4 + 3], w);
    }
}

// ---------------------------------------------------------------------------
// Tail kernel for N % 256 != 0 (not hit for N=524288, kept for generality)
// ---------------------------------------------------------------------------
__global__ void shade_tail_kernel(
    const float* __restrict__ albedos, const float* __restrict__ normals,
    const float* __restrict__ ldirs,   const float* __restrict__ lcols,
    const float* __restrict__ vis,     const float* __restrict__ weights,
    const float* __restrict__ shininess, const float* __restrict__ c2w,
    const int* __restrict__ ray_idx,   float* __restrict__ acc,
    int start, int N, int ngroups_rt)
{
    int t = start + blockIdx.x * blockDim.x + threadIdx.x;
    if (t >= N) return;
    float r0, r1, r2;
    shade_sample_full<0>(t, ngroups_rt, albedos, normals, ldirs, lcols, vis,
                         shininess, c2w, r0, r1, r2);
    float w = weights[t];
    int ray = ray_idx[t];
    atomicAdd(&acc[(size_t)ray * 4 + 0], w * r0);
    atomicAdd(&acc[(size_t)ray * 4 + 1], w * r1);
    atomicAdd(&acc[(size_t)ray * 4 + 2], w * r2);
    atomicAdd(&acc[(size_t)ray * 4 + 3], w);
}

// ---------------------------------------------------------------------------
// Final composite: background blend + sRGB OETF + clamp
// ---------------------------------------------------------------------------
__global__ void composite_kernel(const float* __restrict__ acc,
                                 const float* __restrict__ bg,
                                 float* __restrict__ out, int R)
{
    int r = blockIdx.x * blockDim.x + threadIdx.x;
    if (r >= R) return;
    float aw = acc[(size_t)r * 4 + 3];
#pragma unroll
    for (int c = 0; c < 3; ++c) {
        float x = acc[(size_t)r * 4 + c] + bg[(size_t)r * 3 + c] * (1.0f - aw);
        const float thr = 0.0031308f;
        float y;
        if (x <= thr) {
            y = 12.92f * x;
        } else {
            // powf(max(x,thr), 1/2.4)
            y = 1.055f * exp2f((1.0f / 2.4f) * log2f(fmaxf(x, thr))) - 0.055f;
        }
        out[(size_t)r * 3 + c] = fminf(fmaxf(y, 0.0f), 1.0f);
    }
}

// ---------------------------------------------------------------------------
extern "C" void kernel_launch(void* const* d_in, const int* in_sizes, int n_in,
                              void* d_out, int out_size, void* d_ws, size_t ws_size,
                              hipStream_t stream)
{
    const float* albedos   = (const float*)d_in[0];
    const float* normals   = (const float*)d_in[1];
    const float* ldirs     = (const float*)d_in[2];
    const float* lcols     = (const float*)d_in[3];
    const float* vis       = (const float*)d_in[4];
    const float* bg        = (const float*)d_in[5];
    const float* weights   = (const float*)d_in[6];
    const float* shininess = (const float*)d_in[7];
    const float* c2w       = (const float*)d_in[8];
    const int*   ray_idx   = (const int*)d_in[9];

    const int N = in_sizes[0] / 3;                 // samples
    const int J = in_sizes[2] / in_sizes[0];       // lights: (N*J*3)/(N*3)
    const int R = in_sizes[5] / 3;                 // rays
    const int ngroups = J / 4;

    float* acc = (float*)d_ws;                     // R*4 f32 accumulators

    zero_ws_kernel<<<(R * 4 + 255) / 256, 256, 0, stream>>>(acc, R * 4);

    const int nFullBlocks = N / 256;
    if (nFullBlocks > 0) {
        if (ngroups == 16) {   // J == 64: compile-time specialized, unrolled x4
            shade_kernel<16><<<nFullBlocks, 256, 0, stream>>>(
                albedos, normals, ldirs, lcols, vis, weights, shininess, c2w,
                ray_idx, acc, ngroups);
        } else {
            shade_kernel<0><<<nFullBlocks, 256, 0, stream>>>(
                albedos, normals, ldirs, lcols, vis, weights, shininess, c2w,
                ray_idx, acc, ngroups);
        }
    }
    const int tailStart = nFullBlocks * 256;
    const int tailN = N - tailStart;
    if (tailN > 0) {
        shade_tail_kernel<<<(tailN + 255) / 256, 256, 0, stream>>>(
            albedos, normals, ldirs, lcols, vis, weights, shininess, c2w,
            ray_idx, acc, tailStart, N, ngroups);
    }

    composite_kernel<<<(R + 255) / 256, 256, 0, stream>>>(acc, bg,
                                                          (float*)d_out, R);
}
